// ASAMLayer_12120397709587
// MI455X (gfx1250) — compile-verified, matrix operand order checked by reading, not run on hardware
//
#include <hip/hip_runtime.h>

// ---------------------------------------------------------------------------
// Problem constants (match the JAX reference)
// ---------------------------------------------------------------------------
#define BATCH    2
#define S_LEN    2048
#define D_MODEL  1024
#define NHEAD    16
#define DHEAD    64
#define INNER    1024
#define QKV_LD   3072          // 3*INNER, row stride of the fused qkv buffer
#define NTOK     4096          // B*S
#define FF_DIM   4096
#define WINDOW   128
#define LN_EPS   1e-5f

// LDS tile geometry for the GEMM: 128x32 f16 tiles, rows padded to 40 halves
// (80B = 20 banks) so the 16 same-half lanes hit disjoint 4-bank groups.
#define TILE_ROWS 128
#define TILE_K    32
#define ROW_PAD   40           // halves per padded row
#define ROW_PADB  80           // bytes per padded row

typedef _Float16 v8h  __attribute__((ext_vector_type(8)));
typedef _Float16 v16h __attribute__((ext_vector_type(16)));
typedef float    v8f  __attribute__((ext_vector_type(8)));

// ---------------------------------------------------------------------------
// Fragment loaders for 16x16x32 f16 WMMA (TN layouts).
// A-layout (MxK=16x32) and B-layout (KxN=32x16) are transpose-symmetric on
// CDNA5: lane r (= lane&15) owns row/col r; halves 0..7 hold k = hi*8+0..7,
// halves 8..15 hold k = 16+hi*8+0..7 (hi = lane>>4). Two 16B loads per lane.
// ---------------------------------------------------------------------------
static __device__ __forceinline__ v16h frag_ld(const _Float16* base, int ld,
                                               int row, int k0, int r, int hi) {
  const _Float16* p = base + (size_t)(row + r) * (size_t)ld + k0 + hi * 8;
  v8h lo = *(const v8h*)(p);
  v8h hv = *(const v8h*)(p + 16);
  return __builtin_shufflevector(lo, hv, 0, 1, 2, 3, 4, 5, 6, 7,
                                 8, 9, 10, 11, 12, 13, 14, 15);
}

static __device__ __forceinline__ v16h frag_lds(const _Float16* tile,
                                                int row, int r, int hi) {
  const _Float16* p = tile + (row + r) * ROW_PAD + hi * 8;
  v8h lo = *(const v8h*)(p);       // ds_load_b128
  v8h hv = *(const v8h*)(p + 16);  // ds_load_b128
  return __builtin_shufflevector(lo, hv, 0, 1, 2, 3, 4, 5, 6, 7,
                                 8, 9, 10, 11, 12, 13, 14, 15);
}

static __device__ __forceinline__ v8f wmma_f16(v16h a, v16h b, v8f c) {
  return __builtin_amdgcn_wmma_f32_16x16x32_f16(false, a, false, b,
                                                (short)0, c, false, false);
}

static __device__ __forceinline__ float gelu_exact(float x) {
  return 0.5f * x * (1.0f + erff(x * 0.70710678118654752f));
}

// ---------------------------------------------------------------------------
// CDNA5 async copy: global -> LDS, 16B per lane, tracked by ASYNCcnt.
// lds_off is the LDS byte offset (low 32 bits of a generic shared pointer).
// ---------------------------------------------------------------------------
static __device__ __forceinline__ void async_g2l_b128(unsigned int lds_off,
                                                      const _Float16* g) {
  asm volatile("global_load_async_to_lds_b128 %0, %1, off"
               :
               : "v"(lds_off), "v"(g)
               : "memory");
}

static __device__ __forceinline__ void wait_asynccnt0() {
  asm volatile("s_wait_asynccnt 0x0" ::: "memory");
}

// Stage one 128x32 f16 tile: 512 16B chunks, 2 per thread (256 threads).
static __device__ __forceinline__ void stage_tile(const _Float16* gbase, int ld,
                                                  int row0, int k0,
                                                  unsigned int lds_base, int tid) {
#pragma unroll
  for (int t = 0; t < 2; ++t) {
    int chunk = tid + t * 256;
    int m = chunk >> 2;        // row within tile
    int c = chunk & 3;         // 16B chunk within the 64B row
    const _Float16* g = gbase + (size_t)(row0 + m) * (size_t)ld + k0 + c * 8;
    async_g2l_b128(lds_base + (unsigned)(m * ROW_PADB + c * 16), g);
  }
}

// ---------------------------------------------------------------------------
// f32 -> f16 conversion (weights)
// ---------------------------------------------------------------------------
__global__ __launch_bounds__(256) void cvt_f32_to_f16(const float* __restrict__ src,
                                                      _Float16* __restrict__ dst,
                                                      int n) {
  int i = blockIdx.x * 256 + threadIdx.x;
  if (i < n) dst[i] = (_Float16)src[i];
}

// ---------------------------------------------------------------------------
// LayerNorm over D=1024, one wave32 per token, f16 output
// ---------------------------------------------------------------------------
__global__ __launch_bounds__(256) void layernorm_f16(const float* __restrict__ x,
                                                     const float* __restrict__ g,
                                                     const float* __restrict__ b,
                                                     _Float16* __restrict__ out) {
  int wave = threadIdx.x >> 5;
  int lane = threadIdx.x & 31;
  int tok = blockIdx.x * 8 + wave;
  if (tok >= NTOK) return;
  const float* row = x + (size_t)tok * D_MODEL;

  float vals[32];
  float s = 0.0f;
#pragma unroll
  for (int i = 0; i < 32; ++i) {
    vals[i] = row[lane + i * 32];
    s += vals[i];
  }
#pragma unroll
  for (int off = 16; off > 0; off >>= 1) s += __shfl_xor(s, off, 32);
  float mu = s * (1.0f / (float)D_MODEL);

  float vs = 0.0f;
#pragma unroll
  for (int i = 0; i < 32; ++i) {
    float d = vals[i] - mu;
    vs += d * d;
  }
#pragma unroll
  for (int off = 16; off > 0; off >>= 1) vs += __shfl_xor(vs, off, 32);
  float inv = rsqrtf(vs * (1.0f / (float)D_MODEL) + LN_EPS);

  _Float16* orow = out + (size_t)tok * D_MODEL;
#pragma unroll
  for (int i = 0; i < 32; ++i) {
    int c = lane + i * 32;
    orow[c] = (_Float16)((vals[i] - mu) * inv * g[c] + b[c]);
  }
}

// ---------------------------------------------------------------------------
// TN GEMM:  C[M,N] = A[M,K] * B[N,K]^T   (f16 operands, f32 accumulate)
// Block = 256 threads (8 waves), workgroup tile 128x128, wave tile 32x64
// (2x4 WMMA tiles, 8 accumulators). A/B 128x32 tiles are double-buffered in
// LDS (~40KB of the 320KB WGP pool) and filled with the CDNA5 async copy
// engine (global_load_async_to_lds_b128, ASYNCcnt), overlapping the fill of
// buffer (cur^1) with WMMA work on buffer (cur):
//   s_wait_asynccnt 0  ->  barrier  ->  issue next tile  ->  compute current
// Epilogue: optional bias, exact GELU, residual add; f32 and/or f16 output.
// Extra epilogue path `vt`: for the QKV GEMM, output columns in the V range
// [2*INNER, 3*INNER) are ALSO stored transposed into vt[b][h*64+d][s] with one
// contiguous 16B store per tile (the f32 C-layout holds the column fixed per
// lane and 8 consecutive tokens in VGPRs) -- a free V^T for flash attention.
// ---------------------------------------------------------------------------
__global__ __launch_bounds__(256) void gemm_tn_wmma(
    const _Float16* __restrict__ A, const _Float16* __restrict__ Bw,
    int M, int N, int K,
    const float* __restrict__ bias, const float* __restrict__ resid,
    int do_gelu,
    float* __restrict__ out32, _Float16* __restrict__ out16,
    _Float16* __restrict__ vt) {
  __shared__ _Float16 smem[2][2][TILE_ROWS * ROW_PAD];  // [buf][A/B][tile]

  int tid = threadIdx.x;
  int lane = tid & 31;
  int wave = tid >> 5;
  int r = lane & 15, hi = lane >> 4;
  int wm = wave >> 1;            // 0..3 -> 32-row slabs
  int wn = wave & 1;             // 0..1 -> 64-col slabs
  int mBlk = blockIdx.x * 128;
  int nBlk = blockIdx.y * 128;

  unsigned int aOff[2], bOff[2];
#pragma unroll
  for (int bf = 0; bf < 2; ++bf) {
    aOff[bf] = (unsigned int)(size_t)&smem[bf][0][0];
    bOff[bf] = (unsigned int)(size_t)&smem[bf][1][0];
  }

  v8f acc[8] = {v8f{}, v8f{}, v8f{}, v8f{}, v8f{}, v8f{}, v8f{}, v8f{}};

  // prologue: fill buffer 0
  stage_tile(A, K, mBlk, 0, aOff[0], tid);
  stage_tile(Bw, K, nBlk, 0, bOff[0], tid);

  int nk = K / TILE_K;
  for (int kb = 0; kb < nk; ++kb) {
    int cur = kb & 1;
    wait_asynccnt0();        // our chunks of buf[cur] have landed in LDS
    __syncthreads();         // everyone's chunks have landed
    if (kb + 1 < nk) {       // overlap: fill the other buffer during compute
      stage_tile(A, K, mBlk, (kb + 1) * TILE_K, aOff[cur ^ 1], tid);
      stage_tile(Bw, K, nBlk, (kb + 1) * TILE_K, bOff[cur ^ 1], tid);
    }
    const _Float16* At = &smem[cur][0][0];
    const _Float16* Bt = &smem[cur][1][0];

    v16h a0 = frag_lds(At, wm * 32, r, hi);
    v16h a1 = frag_lds(At, wm * 32 + 16, r, hi);
    v16h b0 = frag_lds(Bt, wn * 64, r, hi);
    v16h b1 = frag_lds(Bt, wn * 64 + 16, r, hi);
    v16h b2 = frag_lds(Bt, wn * 64 + 32, r, hi);
    v16h b3 = frag_lds(Bt, wn * 64 + 48, r, hi);

    acc[0] = wmma_f16(a0, b0, acc[0]);
    acc[1] = wmma_f16(a0, b1, acc[1]);
    acc[2] = wmma_f16(a0, b2, acc[2]);
    acc[3] = wmma_f16(a0, b3, acc[3]);
    acc[4] = wmma_f16(a1, b0, acc[4]);
    acc[5] = wmma_f16(a1, b1, acc[5]);
    acc[6] = wmma_f16(a1, b2, acc[6]);
    acc[7] = wmma_f16(a1, b3, acc[7]);
  }

  // C layout: lane r holds column (tn*16+r); VGPR i holds row (tm*16+hi*8+i)
#pragma unroll
  for (int tm = 0; tm < 2; ++tm) {
#pragma unroll
    for (int tn = 0; tn < 4; ++tn) {
      v8f c = acc[tm * 4 + tn];
      int ncol = nBlk + wn * 64 + tn * 16 + r;
      int mbase = mBlk + wm * 32 + tm * 16 + hi * 8;
      float bn = bias ? bias[ncol] : 0.0f;
      v8h w;
#pragma unroll
      for (int i = 0; i < 8; ++i) {
        float v = c[i] + bn;
        if (do_gelu) v = gelu_exact(v);
        size_t idx = (size_t)(mbase + i) * (size_t)N + ncol;
        if (resid) v += resid[idx];
        if (out32) out32[idx] = v;
        if (out16) out16[idx] = (_Float16)v;
        w[i] = (_Float16)v;
      }
      if (vt && ncol >= 2 * INNER) {
        int hd = ncol - 2 * INNER;          // h*64 + d
        int bb = mbase >> 11;               // batch (token / S_LEN)
        int s = mbase & (S_LEN - 1);        // sequence position (8-aligned)
        *(v8h*)(vt + ((size_t)bb * INNER + hd) * S_LEN + s) = w;
      }
    }
  }
}

// ---------------------------------------------------------------------------
// Local-window flash attention. One wave per (batch, head, 16-query tile).
// Scores are computed TRANSPOSED (S_T = K_tile x Q^T) so that:
//   - softmax state (m, l) is per-lane (lane = query); one shfl_xor(16)
//     combines the two key halves,
//   - exp(S_T) lands exactly in the B-operand layout for O^T = V^T x P,
//     i.e. zero cross-lane data movement between the two WMMA stages.
// V^T comes pre-transposed from the QKV GEMM epilogue, so all three operands
// (Q, K, V^T) load as contiguous 16B fragments. Scale folded into Q once;
// window + seq-bound mask folded into exp.
// ---------------------------------------------------------------------------
__global__ __launch_bounds__(256) void attn_local_wmma(
    const _Float16* __restrict__ qkv, const _Float16* __restrict__ vt,
    _Float16* __restrict__ ao) {
  int lane = threadIdx.x & 31;
  int wv = threadIdx.x >> 5;
  int tile = blockIdx.x * 8 + wv;          // 0 .. B*H*(S/16)-1 = 4095
  int qt = tile & ((S_LEN / 16) - 1);      // query tile within sequence
  int h = (tile >> 7) & (NHEAD - 1);
  int b = tile >> 11;
  int q0 = qt * 16;
  int r = lane & 15, hi = lane >> 4;

  size_t tokBase = (size_t)b * S_LEN;
  const _Float16* Qp = qkv + tokBase * QKV_LD + h * DHEAD;
  const _Float16* Kp = qkv + tokBase * QKV_LD + INNER + h * DHEAD;
  const _Float16* VTp = vt + ((size_t)b * INNER + h * DHEAD) * S_LEN;

  // Q as B-operand (Q^T), pre-scaled by DH^-0.5 = 0.125 (exact in f16)
  v16h qf[2];
#pragma unroll
  for (int kb = 0; kb < 2; ++kb) {
    v16h t = frag_ld(Qp, QKV_LD, q0, kb * 32, r, hi);
#pragma unroll
    for (int i = 0; i < 16; ++i) t[i] = t[i] * (_Float16)0.125f;
    qf[kb] = t;
  }

  float m_run = -1e30f;
  float l_run = 0.0f;
  v8f o[4] = {v8f{}, v8f{}, v8f{}, v8f{}};  // O^T d-tiles: lane = query

  int kstart = q0 - WINDOW; if (kstart < 0) kstart = 0;
  int kend = q0 + 16 + WINDOW; if (kend > S_LEN) kend = S_LEN;
  int qIdx = q0 + r;

  for (int kk = kstart; kk < kend; kk += 32) {
    v8f c0 = v8f{}, c1 = v8f{};
    {
      v16h ka = frag_ld(Kp, QKV_LD, kk, 0, r, hi);
      v16h kb = frag_ld(Kp, QKV_LD, kk, 32, r, hi);
      c0 = wmma_f16(ka, qf[0], c0);
      c0 = wmma_f16(kb, qf[1], c0);
    }
    int k1 = kk + 16;
    int k1row = (k1 + 16 <= S_LEN) ? k1 : (S_LEN - 16);  // in-bounds; masked below
    {
      v16h ka = frag_ld(Kp, QKV_LD, k1row, 0, r, hi);
      v16h kb = frag_ld(Kp, QKV_LD, k1row, 32, r, hi);
      c1 = wmma_f16(ka, qf[0], c1);
      c1 = wmma_f16(kb, qf[1], c1);
    }

#pragma unroll
    for (int i = 0; i < 8; ++i) {
      int key0 = kk + hi * 8 + i;
      int key1 = kk + 16 + hi * 8 + i;
      int d0 = qIdx - key0; if (d0 < 0) d0 = -d0;
      int d1 = qIdx - key1; if (d1 < 0) d1 = -d1;
      if (d0 > WINDOW) c0[i] = -1e30f;
      if (d1 > WINDOW || key1 >= S_LEN) c1[i] = -1e30f;
    }

    float mx = c0[0];
#pragma unroll
    for (int i = 1; i < 8; ++i) mx = fmaxf(mx, c0[i]);
#pragma unroll
    for (int i = 0; i < 8; ++i) mx = fmaxf(mx, c1[i]);
    mx = fmaxf(mx, __shfl_xor(mx, 16, 32));
    float m_new = fmaxf(m_run, mx);
    float alpha = __expf(m_run - m_new);

    v16h pf;  // P^T as B-operand: halves 0..7 <- c0, halves 8..15 <- c1
    float psum = 0.0f;
#pragma unroll
    for (int i = 0; i < 8; ++i) {
      float p = __expf(c0[i] - m_new);
      psum += p;
      pf[i] = (_Float16)p;
    }
#pragma unroll
    for (int i = 0; i < 8; ++i) {
      float p = __expf(c1[i] - m_new);
      psum += p;
      pf[8 + i] = (_Float16)p;
    }
    psum += __shfl_xor(psum, 16, 32);
    l_run = l_run * alpha + psum;
    m_run = m_new;
#pragma unroll
    for (int t = 0; t < 4; ++t) o[t] *= alpha;

    // V^T as A-operand: contiguous fragments from the pre-transposed buffer.
    // Keys past S spill into the 64-element pad; their p is 0.
    v16h vf[4];
#pragma unroll
    for (int t = 0; t < 4; ++t) vf[t] = frag_ld(VTp, S_LEN, t * 16, kk, r, hi);
#pragma unroll
    for (int t = 0; t < 4; ++t) o[t] = wmma_f16(vf[t], pf, o[t]);
  }

  float invl = 1.0f / l_run;
  _Float16* orow = ao + (tokBase + q0 + r) * (size_t)INNER + h * DHEAD;
#pragma unroll
  for (int t = 0; t < 4; ++t) {
    v8h w;
#pragma unroll
    for (int i = 0; i < 8; ++i) w[i] = (_Float16)(o[t][i] * invl);
    *(v8h*)(orow + t * 16 + hi * 8) = w;
  }
}

// ---------------------------------------------------------------------------
// Host-side orchestration
// ---------------------------------------------------------------------------
extern "C" void kernel_launch(void* const* d_in, const int* in_sizes, int n_in,
                              void* d_out, int out_size, void* d_ws, size_t ws_size,
                              hipStream_t stream) {
  const float* x     = (const float*)d_in[0];
  const float* ln1_g = (const float*)d_in[1];
  const float* ln1_b = (const float*)d_in[2];
  const float* w_qkv = (const float*)d_in[3];
  const float* w_out = (const float*)d_in[4];
  const float* b_out = (const float*)d_in[5];
  const float* ln2_g = (const float*)d_in[6];
  const float* ln2_b = (const float*)d_in[7];
  const float* w_ff1 = (const float*)d_in[8];
  const float* b_ff1 = (const float*)d_in[9];
  const float* w_ff2 = (const float*)d_in[10];
  const float* b_ff2 = (const float*)d_in[11];
  float* out = (float*)d_out;

  char* wp = (char*)d_ws;
  auto carve = [&](size_t bytes) {
    void* p = (void*)wp;
    wp += (bytes + 255) & ~(size_t)255;
    return p;
  };
  _Float16* wq_h = (_Float16*)carve((size_t)QKV_LD * D_MODEL * 2);
  _Float16* wo_h = (_Float16*)carve((size_t)D_MODEL * INNER * 2);
  _Float16* w1_h = (_Float16*)carve((size_t)FF_DIM * D_MODEL * 2);
  _Float16* w2_h = (_Float16*)carve((size_t)D_MODEL * FF_DIM * 2);
  _Float16* h1   = (_Float16*)carve((size_t)NTOK * D_MODEL * 2);
  _Float16* qkv  = (_Float16*)carve((size_t)NTOK * QKV_LD * 2);
  _Float16* vtb  = (_Float16*)carve(((size_t)BATCH * INNER * S_LEN + 64) * 2);
  _Float16* ao   = (_Float16*)carve((size_t)NTOK * INNER * 2);
  float*    x1   = (float*)   carve((size_t)NTOK * D_MODEL * 4);
  _Float16* h2   = (_Float16*)carve((size_t)NTOK * D_MODEL * 2);
  _Float16* f1   = (_Float16*)carve((size_t)NTOK * FF_DIM * 2);

  auto cvt = [&](const float* s, _Float16* d, int n) {
    cvt_f32_to_f16<<<(n + 255) / 256, 256, 0, stream>>>(s, d, n);
  };
  cvt(w_qkv, wq_h, QKV_LD * D_MODEL);
  cvt(w_out, wo_h, D_MODEL * INNER);
  cvt(w_ff1, w1_h, FF_DIM * D_MODEL);
  cvt(w_ff2, w2_h, D_MODEL * FF_DIM);

  // LN1
  layernorm_f16<<<NTOK / 8, 256, 0, stream>>>(x, ln1_g, ln1_b, h1);

  // qkv = h1 @ w_qkv^T  (f16 out; V columns additionally stored transposed)
  gemm_tn_wmma<<<dim3(NTOK / 128, QKV_LD / 128), 256, 0, stream>>>(
      h1, wq_h, NTOK, QKV_LD, D_MODEL, nullptr, nullptr, 0, nullptr, qkv, vtb);

  // local-window attention
  attn_local_wmma<<<(BATCH * NHEAD * (S_LEN / 16)) / 8, 256, 0, stream>>>(qkv, vtb, ao);

  // x1 = x + ao @ w_out^T + b_out
  gemm_tn_wmma<<<dim3(NTOK / 128, D_MODEL / 128), 256, 0, stream>>>(
      ao, wo_h, NTOK, D_MODEL, INNER, b_out, x, 0, x1, nullptr, nullptr);

  // LN2
  layernorm_f16<<<NTOK / 8, 256, 0, stream>>>(x1, ln2_g, ln2_b, h2);

  // f1 = gelu(h2 @ w_ff1^T + b_ff1)
  gemm_tn_wmma<<<dim3(NTOK / 128, FF_DIM / 128), 256, 0, stream>>>(
      h2, w1_h, NTOK, FF_DIM, D_MODEL, b_ff1, nullptr, 1, nullptr, f1, nullptr);

  // out = x1 + f1 @ w_ff2^T + b_ff2
  gemm_tn_wmma<<<dim3(NTOK / 128, D_MODEL / 128), 256, 0, stream>>>(
      f1, w2_h, NTOK, D_MODEL, FF_DIM, b_ff2, x1, 0, out, nullptr, nullptr);
}
